// SelfAttentionHead_21887153340971
// MI455X (gfx1250) — compile-verified, matrix operand order checked by reading, not run on hardware
//
#include <hip/hip_runtime.h>
#include <hip/hip_bf16.h>
#include <cstdint>

// Self-attention head, fused flash-attention style, bf16 WMMA w/ f32 accum.
// B=4, S=2048, D=1024, H=64.  d_ws: Q|K|V bf16 [B*S][H] = 3 MB.
// CDNA5 paths: v_wmma_f32_16x16x32_bf16, global_load_async_to_lds_b128,
// s_wait_asynccnt / s_wait_dscnt, wave32 shuffle softmax.

#define BATCH  4
#define SEQ    2048
#define DMODEL 1024
#define HEAD   64

typedef __attribute__((ext_vector_type(16))) __bf16 v16bf;
typedef __attribute__((ext_vector_type(8)))  float  v8f;
typedef __attribute__((ext_vector_type(2)))  __bf16 bf16x2;

// K-offset of VGPR-pair p for 16-bit A/B fragments (ISA 7.12.2, 16-bit A 16x32).
__device__ __forceinline__ int wmma_k0(int p, int half) {
  return ((p & 4) ? 16 : 0) + (half << 3) + ((p & 3) << 1);
}

// Flat LDS address low 32 bits == LDS byte offset (LDS aperture mapping).
__device__ __forceinline__ uint32_t lds_off(const void* p) {
  return (uint32_t)(uintptr_t)p;
}
// Async 16B global -> LDS copy per lane (ASYNCcnt tracked, no VGPR transit).
__device__ __forceinline__ void async_b128(uint32_t lds, const void* g) {
  asm volatile("global_load_async_to_lds_b128 %0, %1, off"
               :: "v"(lds), "v"((uint64_t)(uintptr_t)g) : "memory");
}
__device__ __forceinline__ void wait_async() {
  asm volatile("s_wait_asynccnt 0x0" ::: "memory");
}

// ---------------- Projection: Out[bf16, MxH] = X[f32, MxD] @ W[f32, DxH] ----
// grid (64, 3): blockIdx.y selects q/k/v.  8 waves x 16 rows = 128 rows/block.
__global__ __launch_bounds__(256)
void proj_kernel(const float* __restrict__ Xq, const float* __restrict__ Xk,
                 const float* __restrict__ Xv,
                 const float* __restrict__ Wq, const float* __restrict__ Wk,
                 const float* __restrict__ Wv,
                 __bf16* __restrict__ Oq, __bf16* __restrict__ Ok,
                 __bf16* __restrict__ Ov) {
  __shared__ __bf16 WldsT[HEAD][36];   // W chunk transposed, 18-bank row stride
  __shared__ float  Alds[8][16][34];   // per-wave A staging, 34-dword stride

  const int which = blockIdx.y;
  const float* X = which == 0 ? Xq : which == 1 ? Xk : Xv;
  const float* W = which == 0 ? Wq : which == 1 ? Wk : Wv;
  __bf16* Out    = which == 0 ? Oq : which == 1 ? Ok : Ov;

  const int tid  = threadIdx.x;
  const int wave = tid >> 5;
  const int lane = tid & 31;
  const int m    = lane & 15;
  const int half = lane >> 4;
  const int row0 = blockIdx.x * 128 + wave * 16;

  v8f acc[4] = {};

  for (int k0 = 0; k0 < DMODEL; k0 += 32) {
    // Stage 32xH W chunk, transposed, f32 -> bf16 (coalesced over n).
    for (int idx = tid; idx < 32 * HEAD; idx += 256) {
      int kk = idx >> 6, n = idx & 63;
      WldsT[n][kk] = (__bf16)W[(k0 + kk) * HEAD + n];
    }
    // Async-stage this wave's 16x32 f32 A tile (global row 128B, LDS row 136B).
    {
      uint32_t abase = lds_off(&Alds[wave][0][0]);
      const char* gx = (const char*)(X + (size_t)row0 * DMODEL + k0);
#pragma unroll
      for (int i = 0; i < 4; ++i) {
        int idx = lane + i * 32;            // 128 16B chunks
        int rm = idx >> 3, q = idx & 7;
        async_b128(abase + rm * 136 + q * 16,
                   gx + (size_t)rm * (DMODEL * 4) + q * 16);
      }
      wait_async();
    }
    __syncthreads();

    v16bf a;
#pragma unroll
    for (int p = 0; p < 8; ++p) {
      int kk = wmma_k0(p, half);
      a[2 * p]     = (__bf16)Alds[wave][m][kk];
      a[2 * p + 1] = (__bf16)Alds[wave][m][kk + 1];
    }
#pragma unroll
    for (int t = 0; t < 4; ++t) {
      int n = t * 16 + m;
      v16bf bfrag;
#pragma unroll
      for (int p = 0; p < 8; ++p) {
        int kk = wmma_k0(p, half);
        bf16x2 pr = *(const bf16x2*)&WldsT[n][kk];
        bfrag[2 * p] = pr.x; bfrag[2 * p + 1] = pr.y;
      }
      acc[t] = __builtin_amdgcn_wmma_f32_16x16x32_bf16(
          false, a, false, bfrag, (short)0, acc[t], false, false);
    }
    __syncthreads();
  }

#pragma unroll
  for (int t = 0; t < 4; ++t)
#pragma unroll
    for (int r = 0; r < 8; ++r) {
      int mrow = half * 8 + r;
      Out[(size_t)(row0 + mrow) * HEAD + t * 16 + m] = (__bf16)acc[t][r];
    }
}

// ---------------- Fused attention ------------------------------------------
// Block: 256 threads = 8 waves = 4 wave-pairs; pair p owns 16 query rows,
// even/odd waves of a pair split the key axis, partials merged via LDS.
// Keys interleaved inside a wave's 32-key chunk: key(kt,n) = base + 2n + kt
// (permutation of the reduction axis; mask/P/V all use the same order).
__global__ __launch_bounds__(256)
void attn_kernel(const __bf16* __restrict__ Q, const __bf16* __restrict__ K,
                 const __bf16* __restrict__ V, const uint8_t* __restrict__ mask,
                 float* __restrict__ Out) {
  __shared__ __bf16 Klds[64][68];      // 64-key chunk, padded rows
  __shared__ __bf16 VldsT[HEAD][68];   // V chunk transposed+interleaved
  __shared__ __bf16 Plds[8][16][36];   // per-wave P reshape buffer
  __shared__ float  Macc[4][16][64];   // split-K merge: odd wave's output
  __shared__ float  Mstat[4][2][16];   // split-K merge: odd wave's m_i / l_i

  const int tid  = threadIdx.x;
  const int wave = tid >> 5;
  const int lane = tid & 31;
  const int m    = lane & 15;
  const int half = lane >> 4;
  const int pair = wave >> 1;
  const int wpar = wave & 1;           // which key half of each 64-key chunk
  const int b    = blockIdx.y;
  const int q0   = blockIdx.x * 64 + pair * 16;
  const size_t rowbase = (size_t)(b * SEQ + q0);

  // Q as two A-fragments over the H=64 reduction (2 x K=32).
  v16bf qa[2];
#pragma unroll
  for (int c = 0; c < 2; ++c)
#pragma unroll
    for (int p = 0; p < 8; ++p) {
      int kk = wmma_k0(p, half);
      bf16x2 pr = *(const bf16x2*)(Q + (rowbase + m) * HEAD + c * 32 + kk);
      qa[c][2 * p] = pr.x; qa[c][2 * p + 1] = pr.y;
    }

  v8f acc[4] = {};
  float m_i[8], l_i[8];
#pragma unroll
  for (int r = 0; r < 8; ++r) { m_i[r] = -1e30f; l_i[r] = 0.0f; }

  for (int kc = 0; kc < SEQ; kc += 64) {
    __syncthreads();                   // WAR on LDS tiles from previous chunk
    {  // K: async copy 64 rows x 128B (LDS rows 136B)
      uint32_t kbase = lds_off(&Klds[0][0]);
      const char* gk = (const char*)(K + (size_t)(b * SEQ + kc) * HEAD);
      int rr = tid >> 3, q = tid & 7;
      async_b128(kbase + rr * 136 + q * 16, gk + rr * 128 + q * 16);
      async_b128(kbase + (rr + 32) * 136 + q * 16,
                 gk + (rr + 32) * 128 + q * 16);
    }
    {  // V: transpose into VldsT[h][wp*32 + kt*16 + n], key = wp*32+2n+kt
      int klocal = tid >> 2, hb = (tid & 3) * 16;
      int kin = klocal & 31, wp = klocal >> 5;
      int jpos = wp * 32 + ((kin & 1) << 4) + (kin >> 1);
      const __bf16* vp = V + (size_t)(b * SEQ + kc + klocal) * HEAD + hb;
#pragma unroll
      for (int j = 0; j < 16; ++j) VldsT[hb + j][jpos] = vp[j];
    }
    wait_async();
    __syncthreads();

    // Scores: this wave's keys are kc + wpar*32 + {2n+kt}.
    v8f s[2] = {};
#pragma unroll
    for (int kt = 0; kt < 2; ++kt)
#pragma unroll
      for (int c = 0; c < 2; ++c) {
        v16bf kb;
#pragma unroll
        for (int p = 0; p < 8; ++p) {
          int kk = wmma_k0(p, half);
          bf16x2 pr =
              *(const bf16x2*)&Klds[wpar * 32 + 2 * m + kt][c * 32 + kk];
          kb[2 * p] = pr.x; kb[2 * p + 1] = pr.y;
        }
        s[kt] = __builtin_amdgcn_wmma_f32_16x16x32_bf16(
            false, qa[c], false, kb, (short)0, s[kt], false, false);
      }

    // Scale + mask. Interleave => one u16 covers both tiles' bytes.
#pragma unroll
    for (int r = 0; r < 8; ++r) {
      int mrow = half * 8 + r;
      const uint16_t* mp = (const uint16_t*)(
          mask + (size_t)(b * SEQ + q0 + mrow) * SEQ + kc + wpar * 32);
      uint16_t mv = mp[m];
      s[0][r] = (mv & 0xFF) ? -1e30f : s[0][r] * 0.125f;   // HEAD^-0.5
      s[1][r] = (mv >> 8)   ? -1e30f : s[1][r] * 0.125f;
    }

    // Online softmax; each row lives in one 16-lane half (xor<16 stays inside).
    float p0[8], p1[8], alpha[8];
#pragma unroll
    for (int r = 0; r < 8; ++r) {
      float cm = fmaxf(s[0][r], s[1][r]);
#pragma unroll
      for (int off = 1; off < 16; off <<= 1)
        cm = fmaxf(cm, __shfl_xor(cm, off, 32));
      float mn = fmaxf(m_i[r], cm);
      alpha[r] = __expf(m_i[r] - mn);
      m_i[r] = mn;
      p0[r] = __expf(s[0][r] - mn);
      p1[r] = __expf(s[1][r] - mn);
      float rs = p0[r] + p1[r];
#pragma unroll
      for (int off = 1; off < 16; off <<= 1)
        rs += __shfl_xor(rs, off, 32);
      l_i[r] = l_i[r] * alpha[r] + rs;
    }
#pragma unroll
    for (int t = 0; t < 4; ++t)
#pragma unroll
      for (int r = 0; r < 8; ++r) acc[t][r] *= alpha[r];

    // Reshape P (D layout -> A layout) through wave-private LDS.
#pragma unroll
    for (int r = 0; r < 8; ++r) {
      int mrow = half * 8 + r;
      Plds[wave][mrow][m]      = (__bf16)p0[r];
      Plds[wave][mrow][16 + m] = (__bf16)p1[r];
    }
    asm volatile("s_wait_dscnt 0x0" ::: "memory");  // wave-local LDS ordering

    v16bf pa;
#pragma unroll
    for (int p = 0; p < 8; ++p) {
      int kk = wmma_k0(p, half);
      bf16x2 pr = *(const bf16x2*)&Plds[wave][m][kk];
      pa[2 * p] = pr.x; pa[2 * p + 1] = pr.y;
    }
    // acc += P @ V  (B rows = reduction index j, from this wave's V slice).
#pragma unroll
    for (int t = 0; t < 4; ++t) {
      v16bf vb;
#pragma unroll
      for (int p = 0; p < 8; ++p) {
        int kk = wmma_k0(p, half);
        bf16x2 pr = *(const bf16x2*)&VldsT[t * 16 + m][wpar * 32 + kk];
        vb[2 * p] = pr.x; vb[2 * p + 1] = pr.y;
      }
      acc[t] = __builtin_amdgcn_wmma_f32_16x16x32_bf16(
          false, pa, false, vb, (short)0, acc[t], false, false);
    }
  }

  // ---- Split-K merge: odd wave publishes, even wave combines + stores. ----
  __syncthreads();
  if (wpar == 1) {
#pragma unroll
    for (int t = 0; t < 4; ++t)
#pragma unroll
      for (int r = 0; r < 8; ++r)
        Macc[pair][half * 8 + r][t * 16 + m] = acc[t][r];
#pragma unroll
    for (int r = 0; r < 8; ++r) {      // 16 lanes write same value: benign
      Mstat[pair][0][half * 8 + r] = m_i[r];
      Mstat[pair][1][half * 8 + r] = l_i[r];
    }
  }
  __syncthreads();
  if (wpar == 0) {
    float sA[8], sB[8], inv[8];
#pragma unroll
    for (int r = 0; r < 8; ++r) {
      float mB = Mstat[pair][0][half * 8 + r];
      float lB = Mstat[pair][1][half * 8 + r];
      float mN = fmaxf(m_i[r], mB);
      sA[r] = __expf(m_i[r] - mN);
      sB[r] = __expf(mB - mN);
      inv[r] = 1.0f / (l_i[r] * sA[r] + lB * sB[r]);
    }
#pragma unroll
    for (int t = 0; t < 4; ++t)
#pragma unroll
      for (int r = 0; r < 8; ++r) {
        int mrow = half * 8 + r;
        float o = acc[t][r] * sA[r] + Macc[pair][mrow][t * 16 + m] * sB[r];
        Out[(rowbase + mrow) * HEAD + t * 16 + m] = o * inv[r];
      }
  }
}

extern "C" void kernel_launch(void* const* d_in, const int* in_sizes, int n_in,
                              void* d_out, int out_size, void* d_ws, size_t ws_size,
                              hipStream_t stream) {
  (void)in_sizes; (void)n_in; (void)out_size; (void)ws_size;
  const float*   key_in   = (const float*)d_in[0];
  const float*   query_in = (const float*)d_in[1];
  const float*   value_in = (const float*)d_in[2];
  const uint8_t* mask     = (const uint8_t*)d_in[3];  // jnp.bool_ = 1 byte/elem
  const float*   Wq       = (const float*)d_in[4];
  const float*   Wk       = (const float*)d_in[5];
  const float*   Wv       = (const float*)d_in[6];
  float* out = (float*)d_out;

  __bf16* q_ws = (__bf16*)d_ws;
  __bf16* k_ws = q_ws + (size_t)BATCH * SEQ * HEAD;
  __bf16* v_ws = k_ws + (size_t)BATCH * SEQ * HEAD;

  proj_kernel<<<dim3(BATCH * SEQ / 128, 3), 256, 0, stream>>>(
      query_in, key_in, value_in, Wq, Wk, Wv, q_ws, k_ws, v_ws);

  attn_kernel<<<dim3(SEQ / 64, BATCH), 256, 0, stream>>>(
      q_ws, k_ws, v_ws, mask, out);
}
